// LengthRegulatorlog_68925635166239
// MI455X (gfx1250) — matrix operand order repaired and checked by reading, not compiled, not run on hardware
//
#include <hip/hip_runtime.h>
#include <hip/hip_bf16.h>

// ---------------------------------------------------------------------------
// Problem constants (match reference)
// ---------------------------------------------------------------------------
#define BB   16
#define TT   512
#define EE   256
#define FF   256
#define KK   3
#define MEL  4096
#define KDIM (KK * EE)          // 768 = GEMM K dimension (k*256 + c)
#define LN_EPS 1e-5f

typedef __attribute__((ext_vector_type(16))) _Float16 v16h;
typedef __attribute__((ext_vector_type(8)))  _Float16 v8h;
typedef __attribute__((ext_vector_type(4)))  _Float16 v4h;
typedef __attribute__((ext_vector_type(8)))  float    v8f;

union HFrag { v16h v; v8h h[2]; };

// ---------------------------------------------------------------------------
// Pack conv weight [F][E][K] f32  ->  [F][k*256 + c] f16  (B-matrix friendly)
// ---------------------------------------------------------------------------
__global__ __launch_bounds__(256) void pack_w_kernel(const float* __restrict__ w,
                                                     _Float16* __restrict__ out) {
    int idx = blockIdx.x * 256 + threadIdx.x;          // 0 .. F*E*K-1
    if (idx >= FF * EE * KK) return;
    int f   = idx / (EE * KK);
    int rem = idx - f * (EE * KK);
    int c   = rem / KK;
    int k   = rem - c * KK;
    out[f * KDIM + k * EE + c] = (_Float16)w[idx];
}

// ---------------------------------------------------------------------------
// Fused conv1d(K=3,pad=1) GEMM + bias + LayerNorm + ReLU.
// PASS 1: input x (f32)  -> h1 (f16)
// PASS 2: input h1 (f16) -> linear(256->1) + ReLU -> log_dur (f32)
//
// Grid: B * (T/64) blocks of 128 threads (4 waves).
// Each wave computes a 16x256 output tile via 16 accumulator WMMA tiles,
// K streamed in 24 chunks of 32 with weights staged in LDS.
// B fragments are pipelined in groups of 4 (two 32-VGPR ping-pong buffers)
// and the schedule is pinned with sched_group_barrier so DS-read latency
// overlaps WMMA execution instead of forcing s_wait_dscnt 0 per op.
// ---------------------------------------------------------------------------
template <int PASS>
__global__ __launch_bounds__(128) void conv_ln_kernel(
    const float*    __restrict__ xin_f32,   // PASS1 input [B,T,256]
    const _Float16* __restrict__ xin_f16,   // PASS2 input [B,T,256]
    const _Float16* __restrict__ wT,        // [256][768] f16
    const float*    __restrict__ cbias,     // [256]
    const float*    __restrict__ gam,       // [256]
    const float*    __restrict__ bet,       // [256]
    const float*    __restrict__ lw,        // [256]  (PASS2)
    const float*    __restrict__ lb,        // [1]    (PASS2)
    _Float16*       __restrict__ hout,      // PASS1 output [B,T,256] f16
    float*          __restrict__ dur_out)   // PASS2 output [B,T]
{
    __shared__ _Float16 xtile[66 * 256];    // rows t-1 .. t+64 (33 KB)
    __shared__ _Float16 wchunk[256 * 32];   // [f][kchunk] (16 KB)
    __shared__ float    sg[256], sbe[256], sbias[256], slw[256];

    const int tid   = threadIdx.x;
    const int lane  = tid & 31;
    const int wave  = tid >> 5;             // 0..3 -> M-tile
    const int lm    = lane & 15;
    const int hi    = lane >> 4;

    const int b      = blockIdx.x >> 3;     // T/64 = 8 tiles per batch
    const int tstart = (blockIdx.x & 7) * 64;

    // ---- stage input tile (66 rows x 256 ch, halo = 1 row each side) ----
    for (int q = tid; q < 66 * 64; q += 128) {     // 4 elems per step
        int i  = q >> 6;                            // local storage row
        int c4 = (q & 63) << 2;                     // channel base
        int t  = tstart + i - 1;
        v4h hv;
        if (t >= 0 && t < TT) {
            if (PASS == 1) {
                const float4 v = *(const float4*)&xin_f32[((size_t)(b * TT + t) << 8) + c4];
                hv[0] = (_Float16)v.x; hv[1] = (_Float16)v.y;
                hv[2] = (_Float16)v.z; hv[3] = (_Float16)v.w;
            } else {
                hv = *(const v4h*)&xin_f16[((size_t)(b * TT + t) << 8) + c4];
            }
        } else {
            hv[0] = hv[1] = hv[2] = hv[3] = (_Float16)0.f;
        }
        *(v4h*)&xtile[i * 256 + c4] = hv;
    }
    for (int i = tid; i < 256; i += 128) {
        sg[i]    = gam[i];
        sbe[i]   = bet[i];
        sbias[i] = cbias[i];
        slw[i]   = (PASS == 2) ? lw[i] : 0.f;
    }

    // ---- accumulators: 16 N-tiles x v8f = full 16x256 per wave ----
    v8f acc[16];
    const v8f zacc = {0.f, 0.f, 0.f, 0.f, 0.f, 0.f, 0.f, 0.f};
#pragma unroll
    for (int n = 0; n < 16; ++n) acc[n] = zacc;

    const int row = wave * 16 + lm;         // A-fragment source row (local)

    for (int kk = 0; kk < KDIM / 32; ++kk) {
        __syncthreads();                     // previous wchunk consumers done

        // A fragment per ISA 16-bit A layout (depends only on xtile, which is
        // immutable after staging) -- issue before staging stores so its DS
        // latency hides under the wchunk staging phase.
        HFrag a;
        {
            const int j0 = kk * 32 + hi * 8;     // elements 0..7
            const int j1 = j0 + 16;              // elements 8..15
            a.h[0] = *(const v8h*)&xtile[(row + (j0 >> 8)) * 256 + (j0 & 255)];
            a.h[1] = *(const v8h*)&xtile[(row + (j1 >> 8)) * 256 + (j1 & 255)];
        }

        // stage weight chunk: wchunk[f][e] = wT[f][kk*32+e]
        for (int s = tid; s < 1024; s += 128) {       // 1024 x 16B
            int f = s >> 2;
            int q = (s & 3) << 3;
            *(float4*)&wchunk[f * 32 + q] =
                *(const float4*)&wT[f * KDIM + kk * 32 + q];
        }
        __syncthreads();

        // ---- B fragments: ping-pong groups of 4 (software pipeline) ----
        HFrag bf0[4], bf1[4];
#pragma unroll
        for (int u = 0; u < 4; ++u) {
            const _Float16* wp = &wchunk[(u * 16 + lm) * 32 + hi * 8];
            bf0[u].h[0] = *(const v8h*)wp;
            bf0[u].h[1] = *(const v8h*)(wp + 16);
        }
#pragma unroll
        for (int g = 0; g < 4; ++g) {
            HFrag* cur = (g & 1) ? bf1 : bf0;
            HFrag* nxt = (g & 1) ? bf0 : bf1;
            if (g < 3) {
#pragma unroll
                for (int u = 0; u < 4; ++u) {
                    const _Float16* wp =
                        &wchunk[(((g + 1) * 4 + u) * 16 + lm) * 32 + hi * 8];
                    nxt[u].h[0] = *(const v8h*)wp;
                    nxt[u].h[1] = *(const v8h*)(wp + 16);
                }
            }
#pragma unroll
            for (int u = 0; u < 4; ++u) {
                const int n = g * 4 + u;
                acc[n] = __builtin_amdgcn_wmma_f32_16x16x32_f16(
                    false, a.v, false, cur[u].v, (short)0, acc[n], false, false);
            }
        }

#if __has_builtin(__builtin_amdgcn_sched_group_barrier)
        // Pin the interleave: DS_READ(8), [DS_READ(8), WMMA(4)]x3, WMMA(4)
        // (0x100 = DS read, 0x008 = MFMA/WMMA)
        __builtin_amdgcn_sched_group_barrier(0x100, 8, 0);
        __builtin_amdgcn_sched_group_barrier(0x100, 8, 0);
        __builtin_amdgcn_sched_group_barrier(0x008, 4, 0);
        __builtin_amdgcn_sched_group_barrier(0x100, 8, 0);
        __builtin_amdgcn_sched_group_barrier(0x008, 4, 0);
        __builtin_amdgcn_sched_group_barrier(0x100, 8, 0);
        __builtin_amdgcn_sched_group_barrier(0x008, 4, 0);
        __builtin_amdgcn_sched_group_barrier(0x008, 4, 0);
#endif
    }

    // ---- bias + LayerNorm statistics ----
    // C layout: acc[n][r] holds row (r + hi*8), col (n*16 + lm).
    float s1[8], s2[8];
#pragma unroll
    for (int r = 0; r < 8; ++r) { s1[r] = 0.f; s2[r] = 0.f; }
#pragma unroll
    for (int n = 0; n < 16; ++n) {
        const float bia = sbias[n * 16 + lm];
#pragma unroll
        for (int r = 0; r < 8; ++r) {
            float v = acc[n][r] + bia;
            acc[n][r] = v;
            s1[r] += v;
            s2[r] += v * v;
        }
    }
    // butterfly across the 16 lanes of this half-wave
#pragma unroll
    for (int mask = 1; mask < 16; mask <<= 1) {
#pragma unroll
        for (int r = 0; r < 8; ++r) {
            s1[r] += __shfl_xor(s1[r], mask);
            s2[r] += __shfl_xor(s2[r], mask);
        }
    }
    float mean[8], rstd[8];
#pragma unroll
    for (int r = 0; r < 8; ++r) {
        mean[r] = s1[r] * (1.f / 256.f);
        float var = s2[r] * (1.f / 256.f) - mean[r] * mean[r];
        rstd[r] = rsqrtf(var + LN_EPS);
    }

    // ---- normalize + ReLU, then store h (PASS1) or linear->dur (PASS2) ----
    float lsum[8];
#pragma unroll
    for (int r = 0; r < 8; ++r) lsum[r] = 0.f;

#pragma unroll
    for (int n = 0; n < 16; ++n) {
        const int col = n * 16 + lm;
        const float g = sg[col], be = sbe[col], w = slw[col];
#pragma unroll
        for (int r = 0; r < 8; ++r) {
            float v = (acc[n][r] - mean[r]) * rstd[r] * g + be;
            v = fmaxf(v, 0.f);
            if (PASS == 1) {
                const int tglob = tstart + wave * 16 + r + hi * 8;
                hout[((size_t)(b * TT + tglob) << 8) + col] = (_Float16)v;
            } else {
                lsum[r] += v * w;
            }
        }
    }

    if (PASS == 2) {
        const float lb0 = lb[0];
#pragma unroll
        for (int mask = 1; mask < 16; mask <<= 1)
#pragma unroll
            for (int r = 0; r < 8; ++r) lsum[r] += __shfl_xor(lsum[r], mask);
        if (lm == 0) {
#pragma unroll
            for (int r = 0; r < 8; ++r) {
                const int tglob = tstart + wave * 16 + r + hi * 8;
                dur_out[b * TT + tglob] = fmaxf(lsum[r] + lb0, 0.f);
            }
        }
    }
}

// ---------------------------------------------------------------------------
// Inclusive prefix sum of durations, one block per batch (Hillis-Steele).
// ---------------------------------------------------------------------------
__global__ __launch_bounds__(512) void cumsum_kernel(const int* __restrict__ target,
                                                     int* __restrict__ cum) {
    __shared__ int s[TT];
    const int b = blockIdx.x, t = threadIdx.x;
    s[t] = target[b * TT + t];
    __syncthreads();
    for (int off = 1; off < TT; off <<= 1) {
        int add = (t >= off) ? s[t - off] : 0;
        __syncthreads();
        s[t] += add;
        __syncthreads();
    }
    cum[b * TT + t] = s[t];
}

// ---------------------------------------------------------------------------
// Length-regulator expansion: out[b,m,:] = x[b,t(m),:] or 0.
// 8 frames / block (one wave per frame); 2 x float4 per lane.
// ---------------------------------------------------------------------------
__global__ __launch_bounds__(256) void expand_kernel(const float* __restrict__ x,
                                                     const int* __restrict__ cum,
                                                     float* __restrict__ out) {
    __shared__ int scum[TT];
    const int tid  = threadIdx.x;
    const int b    = blockIdx.x >> 9;            // MEL/8 = 512 blocks per batch
    const int f0   = (blockIdx.x & 511) << 3;    // first frame of this block
    for (int i = tid; i < TT; i += 256) scum[i] = cum[b * TT + i];
    __syncthreads();

    const int wave = tid >> 5;
    const int lane = tid & 31;
    const int m    = f0 + wave;

    // first token with cum[t] > m  (each lane redundantly; LDS-resident)
    int lo = 0, hh = TT;
    while (lo < hh) {
        int mid = (lo + hh) >> 1;
        if (scum[mid] <= m) lo = mid + 1; else hh = mid;
    }
    const bool valid = (lo < TT);

    float4* dst = (float4*)&out[((size_t)(b * MEL + m)) << 8];
    float4 v0, v1;
    if (valid) {
        const float4* src = (const float4*)&x[((size_t)(b * TT + lo)) << 8];
        v0 = src[lane * 2 + 0];
        v1 = src[lane * 2 + 1];
    } else {
        v0 = make_float4(0.f, 0.f, 0.f, 0.f);
        v1 = v0;
    }
    dst[lane * 2 + 0] = v0;
    dst[lane * 2 + 1] = v1;
}

// ---------------------------------------------------------------------------
// Launch
// ---------------------------------------------------------------------------
extern "C" void kernel_launch(void* const* d_in, const int* in_sizes, int n_in,
                              void* d_out, int out_size, void* d_ws, size_t ws_size,
                              hipStream_t stream) {
    const float* x       = (const float*)d_in[0];
    const int*   target  = (const int*)  d_in[1];
    // d_in[2] = mel_max_length (fixed 4096)
    const float* conv1_w = (const float*)d_in[3];
    const float* conv1_b = (const float*)d_in[4];
    const float* ln1_g   = (const float*)d_in[5];
    const float* ln1_b   = (const float*)d_in[6];
    const float* conv2_w = (const float*)d_in[7];
    const float* conv2_b = (const float*)d_in[8];
    const float* ln2_g   = (const float*)d_in[9];
    const float* ln2_b   = (const float*)d_in[10];
    const float* lin_w   = (const float*)d_in[11];
    const float* lin_b   = (const float*)d_in[12];

    float* out_expand = (float*)d_out;                             // [B,MEL,E]
    float* out_dur    = out_expand + (size_t)BB * MEL * EE;        // [B,T]

    // workspace carve-up
    char* ws = (char*)d_ws;
    _Float16* w1h = (_Float16*)(ws);                               // 384 KB
    _Float16* w2h = (_Float16*)(ws + 393216);                      // 384 KB
    _Float16* h1h = (_Float16*)(ws + 786432);                      // 4 MB
    int*      cum = (int*)     (ws + 786432 + 4194304);            // 32 KB

    // 1) pack weights to f16 B-friendly layout
    pack_w_kernel<<<(FF * EE * KK + 255) / 256, 256, 0, stream>>>(conv1_w, w1h);
    pack_w_kernel<<<(FF * EE * KK + 255) / 256, 256, 0, stream>>>(conv2_w, w2h);

    // 2) conv1 + LN1 + ReLU  (x -> h1 f16)
    conv_ln_kernel<1><<<BB * (TT / 64), 128, 0, stream>>>(
        x, nullptr, w1h, conv1_b, ln1_g, ln1_b, nullptr, nullptr, h1h, nullptr);

    // 3) conv2 + LN2 + ReLU + linear + ReLU  (h1 -> log_dur)
    conv_ln_kernel<2><<<BB * (TT / 64), 128, 0, stream>>>(
        nullptr, h1h, w2h, conv2_b, ln2_g, ln2_b, lin_w, lin_b, nullptr, out_dur);

    // 4) duration cumsum
    cumsum_kernel<<<BB, TT, 0, stream>>>(target, cum);

    // 5) length-regulator gather/expand (writes every output element)
    expand_kernel<<<BB * (MEL / 8), 256, 0, stream>>>(x, cum, out_expand);
}
